// DOFEN_10282151707548
// MI455X (gfx1250) — compile-verified, matrix operand order checked by reading, not run on hardware
//
#include <hip/hip_runtime.h>
#include <hip/hip_bf16.h>
#include <math.h>

typedef __attribute__((ext_vector_type(16))) _Float16 v16h;
typedef __attribute__((ext_vector_type(8)))  _Float16 v8h;
typedef __attribute__((ext_vector_type(8)))  float    v8f;

#define B_       4096
#define NCOL     100
#define NCOND    64
#define DD       4
#define TOTAL_   6400
#define NRODT    1600
#define NEST     160
#define NFOREST  100
#define NHID     128
#define NCLASS   10
#define EPSV     1e-5f

// WMMA f16 16x16x32 B-fragment K index for (vgpr v, wave-half):
//   kidx = 2v + (v>=4 ? 8 : 0) + 8*half   (low half of dword = row kidx, high = kidx+1)
__device__ __forceinline__ int frag_kidx(int v, int half) {
    return 2 * v + (v >= 4 ? 8 : 0) + half * 8;
}

// ---------------------------------------------------------------------------
// K1: fused ConditionGeneration + permute + GN1 + conv2 + ReLU + GN2 + conv3.
// One thread per (batch, group-of-4). Produces w[B, N_RODT] only (26 MB).
// ---------------------------------------------------------------------------
__global__ void __launch_bounds__(256)
k1_cond_w(const float* __restrict__ x,   const float* __restrict__ w1,
          const float* __restrict__ b1,  const int*   __restrict__ perm,
          const float* __restrict__ gn1w, const float* __restrict__ gn1b,
          const float* __restrict__ c2w, const float* __restrict__ c2b,
          const float* __restrict__ gn2w, const float* __restrict__ gn2b,
          const float* __restrict__ c3w, const float* __restrict__ c3b,
          float* __restrict__ wout)
{
    int t = blockIdx.x * 256 + threadIdx.x;
    if (t >= B_ * NRODT) return;
    int b = t / NRODT;
    int g = t - b * NRODT;
    int j0 = g * DD;

    float o[4];
#pragma unroll
    for (int i = 0; i < 4; ++i) {
        int p = perm[j0 + i];
        int q = p / NCOL;                 // condition index
        int c = p - q * NCOL;             // column index
        float v = x[b * NCOL + c] * w1[c * NCOND + q] + b1[c * NCOND + q];
        o[i] = 1.0f / (1.0f + __expf(-v));
    }
    float mu = 0.25f * (o[0] + o[1] + o[2] + o[3]);
    float var = 0.f;
#pragma unroll
    for (int i = 0; i < 4; ++i) { float d = o[i] - mu; var += d * d; }
    float rs = rsqrtf(0.25f * var + EPSV);
    float xn[4];
#pragma unroll
    for (int i = 0; i < 4; ++i)
        xn[i] = (o[i] - mu) * rs * gn1w[j0 + i] + gn1b[j0 + i];
    float h[4];
#pragma unroll
    for (int oo = 0; oo < 4; ++oo) {
        float s = c2b[j0 + oo];
#pragma unroll
        for (int i = 0; i < 4; ++i) s += xn[i] * c2w[g * 16 + i * 4 + oo];
        h[oo] = fmaxf(s, 0.f);
    }
    float mu2 = 0.25f * (h[0] + h[1] + h[2] + h[3]);
    float v2 = 0.f;
#pragma unroll
    for (int i = 0; i < 4; ++i) { float d = h[i] - mu2; v2 += d * d; }
    float rs2 = rsqrtf(0.25f * v2 + EPSV);
    float wv = c3b[g];
#pragma unroll
    for (int i = 0; i < 4; ++i)
        wv += ((h[i] - mu2) * rs2 * gn2w[j0 + i] + gn2b[j0 + i]) * c3w[g * 4 + i];
    wout[t] = wv;
}

// ---------------------------------------------------------------------------
// K2: pre-swizzle B matrices into the WMMA 16x16x32 f16 register layout so
// each lane reads its whole fragment as one contiguous 32-byte load.
//   EpP : [f][kb(5)][nt(8)] tiles of [lane(32)][v(8)] dwords  (4 MB)
//   fc1P: [kb(4)][nt(8)] tiles                                (32 KB)
// dword (lane,v) = pack16( B[kidx][nt*16+l16], B[kidx+1][nt*16+l16] )
// ---------------------------------------------------------------------------
__global__ void __launch_bounds__(256)
k2_pack_b(const float* __restrict__ E, const int* __restrict__ swr,
          const float* __restrict__ fc1w,
          _Float16* __restrict__ EpP, _Float16* __restrict__ fc1P)
{
    const int nEp = NFOREST * (NEST / 32) * (NHID / 16) * 256;   // dwords
    const int nFc = (NHID / 32) * (NHID / 16) * 256;             // dwords
    int t = blockIdx.x * 256 + threadIdx.x;
    if (t < nEp) {
        int f  = t / 10240;          // 5*8*256
        int r  = t - f * 10240;
        int kb = r / 2048;  r -= kb * 2048;
        int nt = r / 256;   r -= nt * 256;
        int lane = r >> 3, v = r & 7;
        int half = lane >> 4, l16 = lane & 15;
        int kidx = kb * 32 + frag_kidx(v, half);
        int col  = nt * 16 + l16;
        float lo = E[swr[f * NEST + kidx]     * NHID + col];
        float hi = E[swr[f * NEST + kidx + 1] * NHID + col];
        EpP[2 * t]     = (_Float16)lo;
        EpP[2 * t + 1] = (_Float16)hi;
    } else if (t < nEp + nFc) {
        int u = t - nEp;
        int kb = u / 2048;  u -= kb * 2048;
        int nt = u / 256;   u -= nt * 256;
        int lane = u >> 3, v = u & 7;
        int half = lane >> 4, l16 = lane & 15;
        int kidx = kb * 32 + frag_kidx(v, half);
        int col  = nt * 16 + l16;
        int d = t - nEp;
        fc1P[2 * d]     = (_Float16)fc1w[kidx * NHID + col];
        fc1P[2 * d + 1] = (_Float16)fc1w[(kidx + 1) * NHID + col];
    }
}

// Load a 16x32 A fragment for row m from an LDS row (two contiguous 16B chunks).
__device__ __forceinline__ v16h load_a_frag(const _Float16* row, int kb, int half) {
    const v8h alo = *(const v8h*)(row + kb * 32 + half * 8);
    const v8h ahi = *(const v8h*)(row + kb * 32 + half * 8 + 16);
    return __builtin_shufflevector(alo, ahi, 0, 1, 2, 3, 4, 5, 6, 7,
                                   8, 9, 10, 11, 12, 13, 14, 15);
}

// ---------------------------------------------------------------------------
// K3: forest matmul with fused gather + softmax.
// Block = (b-tile of 16 rows, forest f); 8 waves cover 128 hidden columns.
// F[b,f,:] = softmax_e(w[b, swr[f,e]]) @ Ep[f]   via v_wmma_f32_16x16x32_f16
// B fragments come pre-packed: one 32B load per lane per k-step.
// ---------------------------------------------------------------------------
__global__ void __launch_bounds__(256)
k3_forest_wmma(const float* __restrict__ wbuf, const int* __restrict__ swr,
               const _Float16* __restrict__ EpP, float* __restrict__ Fbuf)
{
    __shared__ float    s_w[16][NEST];
    __shared__ _Float16 s_a[16][NEST];
    const int f   = blockIdx.y;
    const int b0  = blockIdx.x * 16;
    const int tid = threadIdx.x;

    for (int idx = tid; idx < 16 * NEST; idx += 256) {
        int m = idx / NEST, e = idx - m * NEST;
        s_w[m][e] = wbuf[(b0 + m) * NRODT + swr[f * NEST + e]];
    }
    __syncthreads();
    if (tid < 16) {                       // row-wise softmax -> f16 A tile
        float mx = -1e30f;
        for (int e = 0; e < NEST; ++e) mx = fmaxf(mx, s_w[tid][e]);
        float sum = 0.f;
        for (int e = 0; e < NEST; ++e) { float ex = __expf(s_w[tid][e] - mx); s_w[tid][e] = ex; sum += ex; }
        float inv = 1.0f / sum;
        for (int e = 0; e < NEST; ++e) s_a[tid][e] = (_Float16)(s_w[tid][e] * inv);
    }
    __syncthreads();

    const int wave = tid >> 5, lane = tid & 31;
    const int half = lane >> 4, l16 = lane & 15;
    // packed tile base for (f, kb=0, nt=wave): halves
    const _Float16* bp = EpP + ((f * 5) * 8 + wave) * 512 + lane * 16;

    v8f acc = {};
#pragma unroll
    for (int kb = 0; kb < NEST / 32; ++kb) {
        v16h a  = load_a_frag(&s_a[l16][0], kb, half);
        v16h bm = *(const v16h*)(bp + kb * (8 * 512));   // next kb tile: 8 nt * 512 halves
        acc = __builtin_amdgcn_wmma_f32_16x16x32_f16(false, a, false, bm,
                                                     (short)0, acc, false, false);
    }
#pragma unroll
    for (int r = 0; r < 8; ++r) {
        int m = r + half * 8;   // C/D layout: VGPR r -> rows r (lanes 0-15), r+8 (lanes 16-31)
        Fbuf[((b0 + m) * NFOREST + f) * NHID + wave * 16 + l16] = acc[r];
    }
}

// ---------------------------------------------------------------------------
// K4: fused LN1 -> fc1 (WMMA) -> ReLU -> LN2 -> fc2 per 16 (b,f) rows.
// ---------------------------------------------------------------------------
__global__ void __launch_bounds__(256)
k4_mlp(const float* __restrict__ Fbuf, const float* __restrict__ ln1w, const float* __restrict__ ln1b,
       const _Float16* __restrict__ fc1P, const float* __restrict__ fc1b,
       const float* __restrict__ ln2w, const float* __restrict__ ln2b,
       const float* __restrict__ fc2w, const float* __restrict__ fc2b,
       float* __restrict__ ybuf)
{
    __shared__ float    s_f[16][NHID];
    __shared__ _Float16 s_a[16][NHID];
    __shared__ float    s_m[16][NHID];
    const int r0  = blockIdx.x * 16;
    const int tid = threadIdx.x;

    for (int idx = tid; idx < 16 * NHID; idx += 256) {
        int m = idx >> 7, d = idx & 127;
        s_f[m][d] = Fbuf[(r0 + m) * NHID + d];
    }
    __syncthreads();
    if (tid < 16) {                       // LayerNorm 1 -> f16 A tile
        float mu = 0.f;
        for (int d = 0; d < NHID; ++d) mu += s_f[tid][d];
        mu *= (1.0f / NHID);
        float var = 0.f;
        for (int d = 0; d < NHID; ++d) { float dv = s_f[tid][d] - mu; var += dv * dv; }
        float rs = rsqrtf(var * (1.0f / NHID) + EPSV);
        for (int d = 0; d < NHID; ++d)
            s_a[tid][d] = (_Float16)((s_f[tid][d] - mu) * rs * ln1w[d] + ln1b[d]);
    }
    __syncthreads();

    const int wave = tid >> 5, lane = tid & 31;
    const int half = lane >> 4, l16 = lane & 15;
    const _Float16* bp = fc1P + wave * 512 + lane * 16;   // (kb=0, nt=wave)

    v8f acc = {};
#pragma unroll
    for (int kb = 0; kb < NHID / 32; ++kb) {
        v16h a  = load_a_frag(&s_a[l16][0], kb, half);
        v16h bm = *(const v16h*)(bp + kb * (8 * 512));
        acc = __builtin_amdgcn_wmma_f32_16x16x32_f16(false, a, false, bm,
                                                     (short)0, acc, false, false);
    }
    float bias = fc1b[wave * 16 + l16];
#pragma unroll
    for (int r = 0; r < 8; ++r) {
        int m = r + half * 8;
        s_m[m][wave * 16 + l16] = fmaxf(acc[r] + bias, 0.f);   // bias + ReLU
    }
    __syncthreads();
    if (tid < 16) {                       // LayerNorm 2 (in place)
        float mu = 0.f;
        for (int d = 0; d < NHID; ++d) mu += s_m[tid][d];
        mu *= (1.0f / NHID);
        float var = 0.f;
        for (int d = 0; d < NHID; ++d) { float dv = s_m[tid][d] - mu; var += dv * dv; }
        float rs = rsqrtf(var * (1.0f / NHID) + EPSV);
        for (int d = 0; d < NHID; ++d)
            s_m[tid][d] = (s_m[tid][d] - mu) * rs * ln2w[d] + ln2b[d];
    }
    __syncthreads();
    if (tid < 16 * NCLASS) {              // fc2: 128 -> 10 (VALU, tiny)
        int m = tid / NCLASS, c = tid - m * NCLASS;
        float s = fc2b[c];
        for (int d = 0; d < NHID; ++d) s += s_m[m][d] * fc2w[d * NCLASS + c];
        ybuf[(r0 + m) * NCLASS + c] = s;
    }
}

// ---------------------------------------------------------------------------
// K5: deterministic mean over forests.
// ---------------------------------------------------------------------------
__global__ void __launch_bounds__(256)
k5_mean(const float* __restrict__ ybuf, float* __restrict__ out)
{
    int t = blockIdx.x * 256 + threadIdx.x;
    if (t >= B_ * NCLASS) return;
    int b = t / NCLASS, c = t - b * NCLASS;
    float s = 0.f;
    for (int f = 0; f < NFOREST; ++f)
        s += ybuf[(b * NFOREST + f) * NCLASS + c];
    out[t] = s * (1.0f / NFOREST);
}

extern "C" void kernel_launch(void* const* d_in, const int* in_sizes, int n_in,
                              void* d_out, int out_size, void* d_ws, size_t ws_size,
                              hipStream_t stream) {
    const float* x     = (const float*)d_in[0];
    const float* w1    = (const float*)d_in[1];
    const float* b1    = (const float*)d_in[2];
    const int*   perm  = (const int*)  d_in[3];
    const float* gn1w  = (const float*)d_in[4];
    const float* gn1b  = (const float*)d_in[5];
    const float* c2w   = (const float*)d_in[6];
    const float* c2b   = (const float*)d_in[7];
    const float* gn2w  = (const float*)d_in[8];
    const float* gn2b  = (const float*)d_in[9];
    const float* c3w   = (const float*)d_in[10];
    const float* c3b   = (const float*)d_in[11];
    const int*   swr   = (const int*)  d_in[12];
    const float* E     = (const float*)d_in[13];
    const float* ln1w  = (const float*)d_in[14];
    const float* ln1b  = (const float*)d_in[15];
    const float* fc1w  = (const float*)d_in[16];
    const float* fc1b  = (const float*)d_in[17];
    const float* ln2w  = (const float*)d_in[18];
    const float* ln2b  = (const float*)d_in[19];
    const float* fc2w  = (const float*)d_in[20];
    const float* fc2b  = (const float*)d_in[21];
    float* out = (float*)d_out;

    // workspace carve-up (256B aligned)
    char* ws = (char*)d_ws;
    float*    wbuf = (float*)   (ws + 0);           //  26,214,400 B : w[B,1600]
    _Float16* EpP  = (_Float16*)(ws + 26214400);    //   4,096,000 B : Ep packed f16
    _Float16* fc1P = (_Float16*)(ws + 30310400);    //      32,768 B : fc1 packed f16
    float*    Fbuf = (float*)   (ws + 30343168);    // 209,715,200 B : F[B,100,128]
    float*    ybuf = (float*)   (ws + 240058368);   //  16,384,000 B : y[B,100,10]

    // K1: phi_2 fused -> w  (25600 blocks x 256)
    k1_cond_w<<<(B_ * NRODT) / 256, 256, 0, stream>>>(
        x, w1, b1, perm, gn1w, gn1b, c2w, c2b, gn2w, gn2b, c3w, c3b, wbuf);

    // K2: pack B operands into WMMA register layout (f16)
    {
        int n = NFOREST * 5 * 8 * 256 + 4 * 8 * 256;   // dwords
        k2_pack_b<<<(n + 255) / 256, 256, 0, stream>>>(E, swr, fc1w, EpP, fc1P);
    }

    // K3: gather + softmax + forest WMMA  (grid 256 x 100)
    k3_forest_wmma<<<dim3(B_ / 16, NFOREST), 256, 0, stream>>>(wbuf, swr, EpP, Fbuf);

    // K4: LN -> fc1 WMMA -> ReLU -> LN -> fc2  (25600 blocks)
    k4_mlp<<<(B_ * NFOREST) / 16, 256, 0, stream>>>(
        Fbuf, ln1w, ln1b, fc1P, fc1b, ln2w, ln2b, fc2w, fc2b, ybuf);

    // K5: mean over forests
    k5_mean<<<(B_ * NCLASS + 255) / 256, 256, 0, stream>>>(ybuf, out);
}
